// ParzenWindowKernel_82463372083230
// MI455X (gfx1250) — compile-verified, matrix operand order checked by reading, not run on hardware
//
#include <hip/hip_runtime.h>
#include <stdint.h>

typedef __attribute__((ext_vector_type(8))) int v8i;

// H = 0.5 -> predicate |diff| <= 0.25 ; coef/mean = (1/H^32)/4096 = 2^20
#define THRESH 0.25f
#define OUTSCALE 1048576.0f

__global__ __launch_bounds__(128) void
parzen_wmma_kernel(const float* __restrict__ test_Xs,
                   const float* __restrict__ train_Xs,
                   float* __restrict__ out,
                   int chunks)   // M/64
{
    // One 8KB LDS chunk of 64 train rows x 32 dims, shared by 4 waves.
    __shared__ __align__(16) float lds_train[64 * 32];

    const int lane = threadIdx.x & 31;
    const int wave = threadIdx.x >> 5;           // 0..3
    const int tile = blockIdx.x * 4 + wave;      // 16 test rows per wave-tile
    const int row  = tile * 16 + (lane & 15);    // this lane's A-matrix row M
    const int g    = lane >> 4;                  // lane group selects K-halves

    // Pin my test row in registers (32 VGPRs).
    float t[32];
#pragma unroll
    for (int d = 0; d < 32; ++d) t[d] = test_Xs[row * 32 + d];

    v8i acc = {};            // i32 16x16 accumulator (counts per test row)
    v8i bones;               // all-ones u8 B matrix (64x16)
#pragma unroll
    for (int i = 0; i < 8; ++i) bones[i] = 0x01010101;

    for (int c = 0; c < chunks; ++c) {
        __syncthreads();     // previous chunk fully consumed by all waves

        // Async-stage 64 train rows (8KB) into LDS: 512 x b128, 128 threads x 4.
        {
            const float* src = train_Xs + (size_t)c * 64 * 32;
#pragma unroll
            for (int i = 0; i < 4; ++i) {
                int fidx = ((int)threadIdx.x + i * 128) * 4;   // float index
                uint32_t ldsa = (uint32_t)(uintptr_t)(&lds_train[fidx]);
                uint64_t ga   = (uint64_t)(uintptr_t)(src + fidx);
                asm volatile("global_load_async_to_lds_b128 %0, %1, off"
                             :: "v"(ldsa), "v"(ga) : "memory");
            }
            asm volatile("s_wait_asynccnt 0" ::: "memory");
        }
        __syncthreads();     // chunk visible to every wave

        // Build the 16x64 u8 indicator A-matrix for this wave's 16 test rows.
        uint32_t areg[8];
#pragma unroll
        for (int v = 0; v < 8; ++v) {
            uint32_t packed = 0;
#pragma unroll
            for (int b = 0; b < 4; ++b) {
                // ISA 8-bit A layout: K index for (vgpr v, byte b, lane group g)
                const int K = ((v >> 2) << 5) + (((v >> 1) & 1) << 4)
                            + (g << 3) + ((v & 1) << 2) + b;
                const float4* tr4 = reinterpret_cast<const float4*>(&lds_train[K * 32]);
                float mx = 0.0f;   // Chebyshev distance via max-reduction
#pragma unroll
                for (int i = 0; i < 8; ++i) {
                    float4 q = tr4[i];
                    mx = fmaxf(mx, fabsf(t[4 * i + 0] - q.x));
                    mx = fmaxf(mx, fabsf(t[4 * i + 1] - q.y));
                    mx = fmaxf(mx, fabsf(t[4 * i + 2] - q.z));
                    mx = fmaxf(mx, fabsf(t[4 * i + 3] - q.w));
                }
                packed |= (mx <= THRESH ? 1u : 0u) << (8 * b);
            }
            areg[v] = packed;
        }
        v8i a;
#pragma unroll
        for (int v = 0; v < 8; ++v) a[v] = (int)areg[v];

        // acc[m][n] += sum_k A[m][k] * 1  -> per-test-row inside-count
        acc = __builtin_amdgcn_wmma_i32_16x16x64_iu8(
                  /*sgn_a=*/false, a, /*sgn_b=*/false, bones, acc,
                  /*reuse_a=*/false, /*reuse_b=*/false);
    }

    // C layout: VGPR v, lanes 0-15 -> M=v ; lanes 16-31 -> M=v+8. Any N column works.
    if ((lane & 15) == 0) {
        const int base = tile * 16 + g * 8;
#pragma unroll
        for (int v = 0; v < 8; ++v)
            out[base + v] = (float)acc[v] * OUTSCALE;
    }
}

extern "C" void kernel_launch(void* const* d_in, const int* in_sizes, int n_in,
                              void* d_out, int out_size, void* d_ws, size_t ws_size,
                              hipStream_t stream) {
    const float* test_Xs  = (const float*)d_in[0];   // [N,32]
    const float* train_Xs = (const float*)d_in[1];   // [M,32]
    float* out = (float*)d_out;                      // [N]

    const int N = in_sizes[0] / 32;   // 1024
    const int M = in_sizes[1] / 32;   // 4096
    const int chunks = M / 64;        // 64
    const int blocks = N / 64;        // 16 blocks x 4 waves x 16 rows

    hipLaunchKernelGGL(parzen_wmma_kernel, dim3(blocks), dim3(128), 0, stream,
                       test_Xs, train_Xs, out, chunks);
}
